// Grapher_8031588843843
// MI455X (gfx1250) — matrix operand order repaired and checked by reading, not compiled, not run on hardware
//
#include <hip/hip_runtime.h>
#include <math.h>

// ---- CDNA5 WMMA vector types (wave32) ----
typedef __attribute__((ext_vector_type(16))) __bf16 v16bf;
typedef __attribute__((ext_vector_type(8)))  __bf16 v8bf;
typedef __attribute__((ext_vector_type(8)))  float  v8f;

#define BB   64
#define C1   192
#define C2   384
#define NPIX 784
#define KNN  9
#define CN   (C1*NPIX)   // 150528
#define C2N  (C2*NPIX)   // 301056

// =====================================================================
// GEMM + folded-BN (+GELU) (+residual), all operands bf16 in WMMA-friendly
// layouts: A = weights (M x K row-major bf16), B = activations transposed
// (784 x K bf16, row-contiguous). Every fragment is two contiguous 16B
// per-lane global loads (A: chunks at k0+8*half and k0+16+8*half; B: same
// chunks), matching ISA 7.12.2 16-bit A/B layouts. No LDS staging,
// no barriers in the (fully unrolled, KTOT templated) main loop.
// Block: 256 thr = 8 waves; tile 96(M) x 128(N); wave w owns N-tile w.
// grid = (7 n-chunks, batch, z) ; z = m-block (fc1/fc2) or group (gconv).
// OUT_T: write bf16 output transposed (N, outStride) via LDS transpose.
// =====================================================================
template<int KTOT, bool GELU_ACT, bool RESIDUAL, bool OUT_T>
__global__ void __launch_bounds__(256)
gemm_bn_kernel(const __bf16* __restrict__ W, long aGroup,
               const __bf16* __restrict__ actT, int actStride, int bColGroup,
               const float* __restrict__ bias,
               const float* __restrict__ bng, const float* __restrict__ bnb,
               const float* __restrict__ bnm, const float* __restrict__ bnv,
               int chanGroup,
               const float* __restrict__ resid,
               float* __restrict__ outF, long oBatch, long oGroup,
               __bf16* __restrict__ outT, int outStride, int outColGroup)
{
    __shared__ float ldsScale[96];
    __shared__ float ldsShift[96];
    __shared__ __attribute__((aligned(16))) __bf16 ldsT[OUT_T ? 128 * 100 : 1];

    const int tid  = threadIdx.x;
    const int lane = tid & 31;
    const int wave = __builtin_amdgcn_readfirstlane(tid >> 5);
    const int half = lane >> 4;
    const int ln   = lane & 15;
    const int n0   = blockIdx.x * 128;
    const int y    = blockIdx.y;
    const int z    = blockIdx.z;

    const int chanOff = z * chanGroup;
    if (tid < 96) {
        int ch = chanOff + tid;
        float s = bng[ch] * rsqrtf(bnv[ch] + 1e-5f);
        ldsScale[tid] = s;
        ldsShift[tid] = (bias[ch] - bnm[ch]) * s + bnb[ch];
    }
    __syncthreads();

    const __bf16* Wz  = W + (size_t)z * aGroup;
    const __bf16* act = actT + (size_t)y * NPIX * actStride + (size_t)z * bColGroup;

    const int n = n0 + wave * 16 + ln;
    const int nrow = (n < NPIX) ? n : (NPIX - 1);   // clamp; bad cols masked at store
    const __bf16* brow = act + (size_t)nrow * actStride;
    const __bf16* arow0 = Wz + (size_t)ln * KTOT + 8 * half;

    v8f acc[6];
    for (int i = 0; i < 6; ++i)
        for (int r = 0; r < 8; ++r) acc[i][r] = 0.0f;

    union U { v16bf v; v8bf h[2]; };
    constexpr int KSTEPS = KTOT >> 5;
#pragma unroll
    for (int ks = 0; ks < KSTEPS; ++ks) {
        const int k0 = ks * 32;
        U bu;
        bu.h[0] = *(const v8bf*)(brow + k0 + 8 * half);
        bu.h[1] = *(const v8bf*)(brow + k0 + 16 + 8 * half);
#pragma unroll
        for (int mt = 0; mt < 6; ++mt) {
            const __bf16* arow = arow0 + (size_t)(mt * 16) * KTOT + k0;
            U au;
            au.h[0] = *(const v8bf*)arow;
            au.h[1] = *(const v8bf*)(arow + 16);
            acc[mt] = __builtin_amdgcn_wmma_f32_16x16x32_bf16(
                false, au.v, false, bu.v, (short)0, acc[mt], false, false);
        }
    }

    // Epilogue: BN (+GELU) (+residual). C/D layout: M = r + 8*half, N = ln.
    if constexpr (OUT_T) {
        for (int mt = 0; mt < 6; ++mt) {
            for (int r = 0; r < 8; ++r) {
                int ol = mt * 16 + r + 8 * half;
                float val = acc[mt][r] * ldsScale[ol] + ldsShift[ol];
                if (GELU_ACT) val = 0.5f * val * (1.0f + erff(val * 0.70710678118f));
                ldsT[(wave * 16 + ln) * 100 + ol] = (__bf16)val;
            }
        }
        __syncthreads();
        // coalesced transposed write-out: rows (n0+nl), cols z*96..+95
        for (int e = tid; e < 128 * 48; e += 256) {
            int nl = e / 48, cp = e - nl * 48;
            int gn = n0 + nl;
            if (gn < NPIX) {
                unsigned int v = *(const unsigned int*)&ldsT[nl * 100 + cp * 2];
                *(unsigned int*)(outT + ((size_t)y * NPIX + gn) * outStride
                                 + z * outColGroup + cp * 2) = v;
            }
        }
    } else {
        if (n < NPIX) {
            for (int mt = 0; mt < 6; ++mt) {
                for (int r = 0; r < 8; ++r) {
                    int ol = mt * 16 + r + 8 * half;
                    float val = acc[mt][r] * ldsScale[ol] + ldsShift[ol];
                    if (GELU_ACT) val = 0.5f * val * (1.0f + erff(val * 0.70710678118f));
                    size_t o = (size_t)y * oBatch + (size_t)z * oGroup + (size_t)ol * NPIX + n;
                    if (RESIDUAL) val += resid[o];
                    outF[o] = val;
                }
            }
        }
    }
}

// ---- f32 -> bf16 convert (weights prep) ----
__global__ void __launch_bounds__(256)
cvt_kernel(const float* __restrict__ s, __bf16* __restrict__ d, int n)
{
    int i = blockIdx.x * 256 + threadIdx.x;
    if (i < n) d[i] = (__bf16)s[i];
}

// ---- x (B,C,N) f32 -> xT (B,N,C) bf16 (coalesced reads) ----
__global__ void __launch_bounds__(256)
xpose_kernel(const float* __restrict__ src, __bf16* __restrict__ dst)
{
    int t = blockIdx.x * 256 + threadIdx.x;
    if (t >= BB * NPIX) return;
    int b = t / NPIX, n = t - b * NPIX;
    const float* p = src + (size_t)b * CN + n;
    __bf16* q = dst + (size_t)t * C1;
    for (int c = 0; c < C1; ++c) q[c] = (__bf16)p[(size_t)c * NPIX];
}

// ---- per-pixel L2 norm: xt (C,N) f32 -> xnT (N,C) bf16 + xtT (N,C) bf16 ----
__global__ void __launch_bounds__(256)
norm_kernel(const float* __restrict__ xt, __bf16* __restrict__ xnT,
            __bf16* __restrict__ xtT)
{
    int t = blockIdx.x * 256 + threadIdx.x;
    if (t >= BB * NPIX) return;
    int b = t / NPIX, n = t - b * NPIX;
    const float* p = xt + (size_t)b * CN + n;
    float ss = 0.0f;
    for (int c = 0; c < C1; ++c) { float v = p[(size_t)c * NPIX]; ss += v * v; }
    float inv = 1.0f / fmaxf(sqrtf(ss), 1e-12f);
    __bf16* qn = xnT + (size_t)t * C1;
    __bf16* qt = xtT + (size_t)t * C1;
    for (int c = 0; c < C1; ++c) {
        float v = p[(size_t)c * NPIX];
        qt[c] = (__bf16)v;
        qn[c] = (__bf16)(v * inv);
    }
}

// =====================================================================
// KNN: rows normalized => dist = 2-2*dot => top-9 largest dots.
// Block = 16 query rows; WMMA Gram stripe 16x784 -> LDS; 9 argmax passes.
// =====================================================================
__global__ void __launch_bounds__(256)
knn_kernel(const __bf16* __restrict__ xnT, int* __restrict__ nnidx)
{
    __shared__ __attribute__((aligned(32))) float dots[16 * 800];
    const int tid  = threadIdx.x;
    const int lane = tid & 31;
    const int wave = __builtin_amdgcn_readfirstlane(tid >> 5);
    const int half = lane >> 4;
    const int ln   = lane & 15;
    const int m0   = blockIdx.x * 16;
    const int b    = blockIdx.y;
    const __bf16* base = xnT + (size_t)b * NPIX * C1;

    union U { v16bf v; v8bf h[2]; };
    U a[6];
    {
        const __bf16* arow = base + (size_t)(m0 + ln) * C1;
#pragma unroll
        for (int ks = 0; ks < 6; ++ks) {
            a[ks].h[0] = *(const v8bf*)(arow + ks * 32 + 8 * half);
            a[ks].h[1] = *(const v8bf*)(arow + ks * 32 + 16 + 8 * half);
        }
    }
    for (int ct = wave; ct < 49; ct += 8) {
        const __bf16* brow = base + (size_t)(ct * 16 + ln) * C1;
        v8f acc;
        for (int r = 0; r < 8; ++r) acc[r] = 0.0f;
#pragma unroll
        for (int ks = 0; ks < 6; ++ks) {
            U bu;
            bu.h[0] = *(const v8bf*)(brow + ks * 32 + 8 * half);
            bu.h[1] = *(const v8bf*)(brow + ks * 32 + 16 + 8 * half);
            acc = __builtin_amdgcn_wmma_f32_16x16x32_bf16(
                false, a[ks].v, false, bu.v, (short)0, acc, false, false);
        }
        for (int r = 0; r < 8; ++r)
            dots[(r + 8 * half) * 800 + ct * 16 + ln] = acc[r];
    }
    __syncthreads();

    for (int rr = 0; rr < 2; ++rr) {
        int m = wave + rr * 8;
        float* row = &dots[m * 800];
        for (int it = 0; it < KNN; ++it) {
            float bv = -3.4e38f; int bi = 0;
            for (int j = lane; j < NPIX; j += 32) {
                float v = row[j];
                if (v > bv || (v == bv && j < bi)) { bv = v; bi = j; }
            }
            for (int off = 16; off > 0; off >>= 1) {
                float ov = __shfl_xor(bv, off);
                int   oi = __shfl_xor(bi, off);
                if (ov > bv || (ov == bv && oi < bi)) { bv = ov; bi = oi; }
            }
            if (lane == 0) {
                nnidx[((size_t)b * NPIX + m0 + m) * KNN + it] = bi;
                row[bi] = -3.4e38f;
            }
            __syncthreads();
        }
    }
}

// =====================================================================
// Max-relative + interleave, fully coalesced: block = one pixel, 192 thr.
// Every neighbor fetch is a contiguous 384B row of xtT; output row of stT
// (x_c0, mr_c0, x_c1, mr_c1, ...) written as contiguous 4B per thread.
// =====================================================================
__global__ void __launch_bounds__(192)
maxrel_kernel(const __bf16* __restrict__ xtT, const int* __restrict__ nnidx,
              __bf16* __restrict__ stT)
{
    const int n = blockIdx.x, b = blockIdx.y;
    const int c = threadIdx.x;
    const __bf16* base = xtT + (size_t)b * NPIX * C1;
    const int* ip = nnidx + ((size_t)b * NPIX + n) * KNN;
    __bf16 xb = base[(size_t)n * C1 + c];
    float xf = (float)xb;
    float best = -3.4e38f;
    for (int k = 0; k < KNN; ++k) {
        float v = (float)base[(size_t)ip[k] * C1 + c];
        best = fmaxf(best, v);
    }
    __bf16* orow = stT + ((size_t)b * NPIX + n) * C2;
    orow[2 * c]     = xb;
    orow[2 * c + 1] = (__bf16)(best - xf);
}

extern "C" void kernel_launch(void* const* d_in, const int* in_sizes, int n_in,
                              void* d_out, int out_size, void* d_ws, size_t ws_size,
                              hipStream_t stream)
{
    const float* x     = (const float*)d_in[0];
    const float* fc1_w = (const float*)d_in[1];
    const float* fc1_b = (const float*)d_in[2];
    const float* bn1_g = (const float*)d_in[3];
    const float* bn1_b = (const float*)d_in[4];
    const float* bn1_m = (const float*)d_in[5];
    const float* bn1_v = (const float*)d_in[6];
    const float* gc_w  = (const float*)d_in[7];
    const float* gc_b  = (const float*)d_in[8];
    const float* bn2_g = (const float*)d_in[9];
    const float* bn2_b = (const float*)d_in[10];
    const float* bn2_m = (const float*)d_in[11];
    const float* bn2_v = (const float*)d_in[12];
    const float* fc2_w = (const float*)d_in[13];
    const float* fc2_b = (const float*)d_in[14];
    const float* bn3_g = (const float*)d_in[15];
    const float* bn3_b = (const float*)d_in[16];
    const float* bn3_m = (const float*)d_in[17];
    const float* bn3_v = (const float*)d_in[18];

    // workspace carve (~176 MB)
    char* w = (char*)d_ws;
    float*  xt    = (float*)w;   w += (size_t)BB * CN * 4;          // fc1 out (C,N) f32
    __bf16* xTbf  = (__bf16*)w;  w += (size_t)BB * CN * 2;          // x  (N,C) bf16
    __bf16* xnT   = (__bf16*)w;  w += (size_t)BB * CN * 2;          // normalized (N,C)
    __bf16* xtT   = (__bf16*)w;  w += (size_t)BB * CN * 2;          // fc1 out (N,C) bf16
    int*    nnidx = (int*)w;     w += (size_t)BB * NPIX * KNN * 4;
    __bf16* stT   = (__bf16*)w;  w += (size_t)BB * NPIX * C2 * 2;   // (N,384)
    __bf16* gT    = (__bf16*)w;  w += (size_t)BB * NPIX * C2 * 2;   // (N,384)
    __bf16* w1b   = (__bf16*)w;  w += (size_t)C1 * C1 * 2;          // fc1_w bf16
    __bf16* wgb   = (__bf16*)w;  w += (size_t)4 * 96 * 96 * 2;      // gc_w bf16
    __bf16* w2b   = (__bf16*)w;  w += (size_t)C1 * C2 * 2;          // fc2_w bf16

    dim3 blk(256);

    // Prep: weights -> bf16 ; x -> (N,C) bf16
    cvt_kernel<<<dim3((C1*C1 + 255)/256), blk, 0, stream>>>(fc1_w, w1b, C1*C1);
    cvt_kernel<<<dim3((4*96*96 + 255)/256), blk, 0, stream>>>(gc_w, wgb, 4*96*96);
    cvt_kernel<<<dim3((C1*C2 + 255)/256), blk, 0, stream>>>(fc2_w, w2b, C1*C2);
    xpose_kernel<<<dim3((BB*NPIX + 255)/256), blk, 0, stream>>>(x, xTbf);

    // K1: fc1 + BN1 -> xt (C,N) f32
    gemm_bn_kernel<C1, false, false, false><<<dim3(7, BB, 2), blk, 0, stream>>>(
        w1b, (long)96 * C1,
        xTbf, C1, 0,
        fc1_b, bn1_g, bn1_b, bn1_m, bn1_v, 96,
        nullptr,
        xt, (long)CN, (long)96 * NPIX,
        nullptr, 0, 0);

    // K2a: normalize -> xnT + bf16 copy xtT (both N-major)
    norm_kernel<<<dim3((BB*NPIX + 255)/256), blk, 0, stream>>>(xt, xnT, xtT);

    // K2b: WMMA Gram + top-9
    knn_kernel<<<dim3(49, BB), blk, 0, stream>>>(xnT, nnidx);

    // K3: max-relative -> stT (N,384) bf16
    maxrel_kernel<<<dim3(NPIX, BB), dim3(192), 0, stream>>>(xtT, nnidx, stT);

    // K4: grouped conv + BN2 + GELU -> gT (N,384) bf16 (transposed epilogue)
    gemm_bn_kernel<96, true, false, true><<<dim3(7, BB, 4), blk, 0, stream>>>(
        wgb, (long)96 * 96,
        stT, C2, 96,
        gc_b, bn2_g, bn2_b, bn2_m, bn2_v, 96,
        nullptr,
        nullptr, 0L, 0L,
        gT, C2, 96);

    // K5: fc2 + BN3 + residual -> d_out (C,N) f32
    gemm_bn_kernel<C2, false, true, false><<<dim3(7, BB, 2), blk, 0, stream>>>(
        w2b, (long)96 * C2,
        gT, C2, 0,
        fc2_b, bn3_g, bn3_b, bn3_m, bn3_v, 96,
        x,
        (float*)d_out, (long)CN, (long)96 * NPIX,
        nullptr, 0, 0);
}
